// FPDIoULoss_73074573574347
// MI455X (gfx1250) — compile-verified, matrix operand order checked by reading, not run on hardware
//
#include <hip/hip_runtime.h>
#include <hip/hip_fp16.h>
#include <math.h>

typedef __attribute__((ext_vector_type(16))) _Float16 v16h;
typedef __attribute__((ext_vector_type(8)))  float    v8f;

// ---------------------------------------------------------------------------
// Block-wide sum of 256 f32 values using one v_wmma_f32_16x16x32_f16.
// Each value v is split v = hi + lo (two f16, error ~v*2^-22). The 512 f16
// values fill the 16x32 A matrix (any packing works: with B == ones,
// D[m][n] = rowsum(A,m), so sum over all 256 D elements = 16 * sum(A)).
// Returns the block total in lane/thread 0 (garbage elsewhere).
// ---------------------------------------------------------------------------
__device__ __forceinline__ float block_sum_wmma(float v, _Float16* sh) {
    const int tid = threadIdx.x;
    _Float16 hi = (_Float16)v;
    _Float16 lo = (_Float16)(v - (float)hi);
    sh[2 * tid]     = hi;
    sh[2 * tid + 1] = lo;
    __syncthreads();
    float s = 0.f;
    if (tid < 32) {                       // wave 0: EXEC all-ones for WMMA
        v16h a = ((const v16h*)sh)[tid];  // 16 f16 per lane -> 512 total
        v16h ones;
        #pragma unroll
        for (int q = 0; q < 16; ++q) ones[q] = (_Float16)1.0f;
        v8f c = {};
        c = __builtin_amdgcn_wmma_f32_16x16x32_f16(
                /*neg_a=*/false, a, /*neg_b=*/false, ones,
                /*c_mod=*/(short)0, c, /*reuse_a=*/false, /*reuse_b=*/false);
        s = c[0] + c[1] + c[2] + c[3] + c[4] + c[5] + c[6] + c[7];
        s += __shfl_xor(s, 16, 32);
        s += __shfl_xor(s, 8, 32);
        s += __shfl_xor(s, 4, 32);
        s += __shfl_xor(s, 2, 32);
        s += __shfl_xor(s, 1, 32);
        s *= 0.0625f;                     // undo 16x replication across N
    }
    return s;
}

// ---------------------------------------------------------------------------
// Per-box FPDIoU loss (forward), mirroring the JAX reference in f32.
// ---------------------------------------------------------------------------
__device__ __forceinline__ float fpdiou_one(const float b1[5], const float b2[5]) {
    const float DXS[4] = {0.5f, -0.5f, -0.5f, 0.5f};
    const float DYS[4] = {0.5f,  0.5f, -0.5f, -0.5f};

    float c1x[4], c1y[4], c2x[4], c2y[4];
    {
        float co = cosf(b1[4]), si = sinf(b1[4]);
        #pragma unroll
        for (int q = 0; q < 4; ++q) {
            float dx = DXS[q] * b1[2], dy = DYS[q] * b1[3];
            c1x[q] = dx * co - dy * si + b1[0];
            c1y[q] = dx * si + dy * co + b1[1];
        }
        co = cosf(b2[4]); si = sinf(b2[4]);
        #pragma unroll
        for (int q = 0; q < 4; ++q) {
            float dx = DXS[q] * b2[2], dy = DYS[q] * b2[3];
            c2x[q] = dx * co - dy * si + b2[0];
            c2y[q] = dx * si + dy * co + b2[1];
        }
    }

    // vertex pool: 0-3 = c1, 4-7 = c2, 8-23 = edge intersections
    float vx[24], vy[24];
    bool  vm[24];
    #pragma unroll
    for (int q = 0; q < 4; ++q) {
        vx[q] = c1x[q]; vy[q] = c1y[q];
        vx[4 + q] = c2x[q]; vy[4 + q] = c2y[q];
    }

    // c1 corners inside box2
    {
        float ax = c2x[0], ay = c2y[0];
        float abx = c2x[1] - ax, aby = c2y[1] - ay;
        float adx = c2x[3] - ax, ady = c2y[3] - ay;
        float abab = abx * abx + aby * aby;
        float adad = adx * adx + ady * ady;
        #pragma unroll
        for (int q = 0; q < 4; ++q) {
            float amx = c1x[q] - ax, amy = c1y[q] - ay;
            float pab = (abx * amx + aby * amy) / abab;
            float pad = (adx * amx + ady * amy) / adad;
            vm[q] = (pab > -1e-6f) && (pab < 1.f + 1e-6f) &&
                    (pad > -1e-6f) && (pad < 1.f + 1e-6f);
        }
    }
    // c2 corners inside box1
    {
        float ax = c1x[0], ay = c1y[0];
        float abx = c1x[1] - ax, aby = c1y[1] - ay;
        float adx = c1x[3] - ax, ady = c1y[3] - ay;
        float abab = abx * abx + aby * aby;
        float adad = adx * adx + ady * ady;
        #pragma unroll
        for (int q = 0; q < 4; ++q) {
            float amx = c2x[q] - ax, amy = c2y[q] - ay;
            float pab = (abx * amx + aby * amy) / abab;
            float pad = (adx * amx + ady * amy) / adad;
            vm[4 + q] = (pab > -1e-6f) && (pab < 1.f + 1e-6f) &&
                        (pad > -1e-6f) && (pad < 1.f + 1e-6f);
        }
    }

    // all 4x4 edge-pair segment intersections
    #pragma unroll
    for (int e1 = 0; e1 < 4; ++e1) {
        float x1 = c1x[e1], y1 = c1y[e1];
        float x2 = c1x[(e1 + 1) & 3], y2 = c1y[(e1 + 1) & 3];
        #pragma unroll
        for (int e2 = 0; e2 < 4; ++e2) {
            float x3 = c2x[e2], y3 = c2y[e2];
            float x4 = c2x[(e2 + 1) & 3], y4 = c2y[(e2 + 1) & 3];
            float num  = (y4 - y3) * (x2 - x1) - (x4 - x3) * (y2 - y1);
            float dent = (x4 - x3) * (y1 - y3) - (y4 - y3) * (x1 - x3);
            float denu = (x2 - x1) * (y1 - y3) - (y2 - y1) * (x1 - x3);
            bool zero = (num == 0.f);
            float t = zero ? -1.f :  dent / num;
            float u = zero ? -1.f : -denu / num;
            bool m = (t > 0.f) && (t < 1.f) && (u > 0.f) && (u < 1.f);
            float t2 = dent / (num + 1e-8f);
            int idx = 8 + e1 * 4 + e2;
            vx[idx] = m ? (x1 + t2 * (x2 - x1)) : 0.f;
            vy[idx] = m ? (y1 + t2 * (y2 - y1)) : 0.f;
            vm[idx] = m;
        }
    }

    // count + masked mean
    int kcnt = 0;
    float sx = 0.f, sy = 0.f;
    #pragma unroll
    for (int q = 0; q < 24; ++q) {
        kcnt += vm[q] ? 1 : 0;
        sx   += vm[q] ? vx[q] : 0.f;
        sy   += vm[q] ? vy[q] : 0.f;
    }
    float kk = (float)(kcnt < 1 ? 1 : kcnt);
    float mx = sx / kk, my = sy / kk;

    // angles (1e9 for masked-out, INF for padding slots)
    float ag[32], ax_[32], ay_[32];
    #pragma unroll
    for (int q = 0; q < 24; ++q) {
        ag[q]  = vm[q] ? atan2f(vy[q] - my, vx[q] - mx) : 1e9f;
        ax_[q] = vx[q];
        ay_[q] = vy[q];
    }
    #pragma unroll
    for (int q = 24; q < 32; ++q) { ag[q] = __builtin_inff(); ax_[q] = 0.f; ay_[q] = 0.f; }

    // fully-unrolled bitonic sort (32 elems) by angle — compile-time indices
    // keep everything in registers (no scratch traffic)
    #pragma unroll
    for (int kb = 2; kb <= 32; kb <<= 1) {
        #pragma unroll
        for (int j = kb >> 1; j > 0; j >>= 1) {
            #pragma unroll
            for (int i = 0; i < 32; ++i) {
                int l = i ^ j;
                if (l > i) {
                    bool up = ((i & kb) == 0);
                    bool gt = (ag[i] > ag[l]);
                    bool sw = (gt == up);
                    float tf;
                    tf = sw ? ag[l]  : ag[i];  ag[l]  = sw ? ag[i]  : ag[l];  ag[i]  = tf;
                    tf = sw ? ax_[l] : ax_[i]; ax_[l] = sw ? ax_[i] : ax_[l]; ax_[i] = tf;
                    tf = sw ? ay_[l] : ay_[i]; ay_[l] = sw ? ay_[i] : ay_[l]; ay_[i] = tf;
                }
            }
        }
    }

    // shoelace over the first kcnt (angle-sorted) vertices
    float x0 = ax_[0], y0 = ay_[0];
    float cs = 0.f;
    #pragma unroll
    for (int q = 0; q < 24; ++q) {
        bool valid = (q < kcnt);
        bool wrap  = (q + 1 >= kcnt);
        float nx = wrap ? x0 : ax_[q + 1];
        float ny = wrap ? y0 : ay_[q + 1];
        float cr = ax_[q] * ny - ay_[q] * nx;
        cs += valid ? cr : 0.f;
    }
    float inter = fabsf(cs) * 0.5f;

    float area1 = b1[2] * b1[3];
    float area2 = b2[2] * b2[3];
    float iou = inter / (area1 + area2 - inter);
    iou = fmaxf(iou, 1e-6f);

    // FPDIoU corner-distance term: sort both corner sets by x (5-CE network)
    float psx[4], psy[4], gsx[4], gsy[4];
    #pragma unroll
    for (int q = 0; q < 4; ++q) { psx[q] = c1x[q]; psy[q] = c1y[q]; gsx[q] = c2x[q]; gsy[q] = c2y[q]; }
    #define CE4(X, Y, a, b)                                                    \
        { bool sw = (X[a] > X[b]); float tf;                                   \
          tf = sw ? X[b] : X[a]; X[b] = sw ? X[a] : X[b]; X[a] = tf;           \
          tf = sw ? Y[b] : Y[a]; Y[b] = sw ? Y[a] : Y[b]; Y[a] = tf; }
    CE4(psx, psy, 0, 1) CE4(psx, psy, 2, 3) CE4(psx, psy, 0, 2)
    CE4(psx, psy, 1, 3) CE4(psx, psy, 1, 2)
    CE4(gsx, gsy, 0, 1) CE4(gsx, gsy, 2, 3) CE4(gsx, gsy, 0, 2)
    CE4(gsx, gsy, 1, 3) CE4(gsx, gsy, 1, 2)
    #undef CE4

    float d0 = gsx[0] - psx[0];
    float d1x = psx[1] - gsx[1], d1y = psy[1] - gsy[1];
    float d2x = psx[2] - gsx[2], d2y = psy[2] - gsy[2];
    float d3x = psx[3] - gsx[3], d3y = psy[3] - gsy[3];
    float d = 2.0f * d0 * d0 + d1x * d1x + d1y * d1y + d2x * d2x + d2y * d2y +
              d3x * d3x + d3y * d3y;
    float res = d * (1.0f / (4.0f * 1024.0f * 1024.0f));

    return 1.0f - iou + res;
}

// ---------------------------------------------------------------------------
// Kernel 1: per-box loss + WMMA block reduction -> one partial per block
// ---------------------------------------------------------------------------
__global__ __launch_bounds__(256)
void fpdiou_partial_kernel(const float* __restrict__ pred,
                           const float* __restrict__ targ,
                           float* __restrict__ bsum, int n) {
    int i = blockIdx.x * 256 + threadIdx.x;
    float loss = 0.f;
    if (i < n) {
        float b1[5], b2[5];
        #pragma unroll
        for (int q = 0; q < 5; ++q) { b1[q] = pred[5 * i + q]; b2[q] = targ[5 * i + q]; }
        loss = fpdiou_one(b1, b2);
    }
    __shared__ _Float16 sh[512] __attribute__((aligned(32)));
    float s = block_sum_wmma(loss, sh);
    if (threadIdx.x == 0) bsum[blockIdx.x] = s;
}

// ---------------------------------------------------------------------------
// Kernel 2: deterministic final reduction of block partials -> mean
// ---------------------------------------------------------------------------
__global__ __launch_bounds__(256)
void fpdiou_final_kernel(const float* __restrict__ bsum, int nb, int n,
                         float* __restrict__ out) {
    float v = 0.f;
    for (int j = threadIdx.x; j < nb; j += 256) v += bsum[j];
    __shared__ _Float16 sh[512] __attribute__((aligned(32)));
    float s = block_sum_wmma(v, sh);
    if (threadIdx.x == 0) out[0] = s / (float)n;
}

extern "C" void kernel_launch(void* const* d_in, const int* in_sizes, int n_in,
                              void* d_out, int out_size, void* d_ws, size_t ws_size,
                              hipStream_t stream) {
    const float* pred = (const float*)d_in[0];
    const float* targ = (const float*)d_in[1];
    float* out = (float*)d_out;
    float* ws  = (float*)d_ws;

    int n  = in_sizes[0] / 5;
    int nb = (n + 255) / 256;

    fpdiou_partial_kernel<<<nb, 256, 0, stream>>>(pred, targ, ws, n);
    fpdiou_final_kernel<<<1, 256, 0, stream>>>(ws, nb, n, out);
}
